// TrueFastGCN_69827578298666
// MI455X (gfx1250) — matrix operand order repaired and checked by reading, not compiled, not run on hardware
//
#include <hip/hip_runtime.h>

// ---------------- constants ----------------
#define NN 100000       // N_NODES
#define NE 3200000      // N_EDGES
#define INF_ 512        // IN_F
#define HD  256         // HID
#define SMP 32768       // SAMPLE
#define BN_EPS 1e-5f

typedef __bf16 bf16;
typedef __attribute__((ext_vector_type(16))) bf16  v16bf;
typedef __attribute__((ext_vector_type(8)))  bf16  v8bf;
typedef __attribute__((ext_vector_type(4)))  bf16  v4bf;
typedef __attribute__((ext_vector_type(8)))  float v8f;
typedef __attribute__((ext_vector_type(4)))  unsigned int v4u;
typedef __attribute__((ext_vector_type(8)))  int   v8i;
typedef __attribute__((ext_vector_type(4)))  int   v4i;

// ---------------- utility fills ----------------
__global__ void fg_fill_f32(float* p, long n, float v) {
  long i = (long)blockIdx.x * blockDim.x + threadIdx.x;
  long g = (long)gridDim.x * blockDim.x;
  for (; i < n; i += g) p[i] = v;
}
__global__ void fg_fill_i32(int* p, long n, int v) {
  long i = (long)blockIdx.x * blockDim.x + threadIdx.x;
  long g = (long)gridDim.x * blockDim.x;
  for (; i < n; i += g) p[i] = v;
}

// ---------------- degree ----------------
__global__ void fg_degree(const int* __restrict__ src, float* __restrict__ deg) {
  long i = (long)blockIdx.x * blockDim.x + threadIdx.x;
  long g = (long)gridDim.x * blockDim.x;
  for (; i < NE; i += g) atomicAdd(&deg[src[i]], 1.0f);
}

// ---------------- inclusive scan (single block, chunked) ----------------
__global__ void fg_scan(const float* __restrict__ deg, float* __restrict__ cdf,
                        float* __restrict__ total) {
  __shared__ float buf[1024];
  __shared__ float carry;
  int tid = threadIdx.x;
  if (tid == 0) carry = 0.0f;
  __syncthreads();
  for (int base = 0; base < NN; base += 1024) {
    int i = base + tid;
    buf[tid] = (i < NN) ? deg[i] : 0.0f;
    __syncthreads();
    for (int off = 1; off < 1024; off <<= 1) {
      float t = (tid >= off) ? buf[tid - off] : 0.0f;
      __syncthreads();
      buf[tid] += t;
      __syncthreads();
    }
    float cprev = carry;
    if (i < NN) cdf[i] = buf[tid] + cprev;
    __syncthreads();
    if (tid == 0) carry = cprev + buf[1023];
    __syncthreads();
  }
  if (tid == 0) *total = carry;
}

// ---------------- sampling: hash RNG + inverse-CDF binary search ----------------
__device__ __forceinline__ unsigned long long fg_mix(unsigned long long x) {
  x += 0x9E3779B97f4A7C15ull;
  x = (x ^ (x >> 30)) * 0xBF58476D1CE4E5B9ull;
  x = (x ^ (x >> 27)) * 0x94D049BB133111EBull;
  return x ^ (x >> 31);
}
__global__ void fg_sample(const float* __restrict__ cdf, const float* __restrict__ total,
                          int* __restrict__ sampled) {
  int i = blockIdx.x * blockDim.x + threadIdx.x;
  if (i >= SMP) return;
  unsigned long long h = fg_mix(0x2Aull * 0x100000001ull + (unsigned long long)i);
  double u = (double)(h >> 11) * (1.0 / 9007199254740992.0);  // [0,1)
  float r = (float)(u * (double)(*total));
  int lo = 0, hi = NN - 1;
  while (lo < hi) {
    int mid = (lo + hi) >> 1;
    if (cdf[mid] > r) hi = mid; else lo = mid + 1;
  }
  sampled[i] = lo;
}

// ---------------- relabel map: last occurrence wins == atomicMax of positions ----
__global__ void fg_mapping(const int* __restrict__ sampled, int* __restrict__ mapping) {
  int i = blockIdx.x * blockDim.x + threadIdx.x;
  if (i >= SMP) return;
  atomicMax(&mapping[sampled[i]], i);
}

// ---------------- valid-edge compaction ----------------
__global__ void fg_compact(const int* __restrict__ src, const int* __restrict__ dst,
                           const int* __restrict__ mapping,
                           int* __restrict__ rowc, int* __restrict__ colc,
                           int* __restrict__ ec) {
  long i = (long)blockIdx.x * blockDim.x + threadIdx.x;
  long g = (long)gridDim.x * blockDim.x;
  for (; i < NE; i += g) {
    int rm = mapping[src[i]];
    int cm = mapping[dst[i]];
    if (rm >= 0 && cm >= 0) {
      int p = atomicAdd(ec, 1);
      rowc[p] = rm;
      colc[p] = cm;
    }
  }
}

// ---------------- gather + f32->bf16 convert: xsb[i,:] = bf16(x[sampled[i],:]) --
__global__ void fg_gather_x(const float* __restrict__ x, const int* __restrict__ sampled,
                            bf16* __restrict__ xsb) {
  long n4 = (long)SMP * (INF_ / 4);
  long i = (long)blockIdx.x * blockDim.x + threadIdx.x;
  long g = (long)gridDim.x * blockDim.x;
  for (; i < n4; i += g) {
    int r = (int)(i >> 7);           // INF_/4 = 128 chunks per row
    int k4 = (int)(i & 127) * 4;
    const float4 v = *(const float4*)(x + (long)sampled[r] * INF_ + k4);
    v4bf o = {(bf16)v.x, (bf16)v.y, (bf16)v.z, (bf16)v.w};
    *(v4bf*)(xsb + (long)r * INF_ + k4) = o;
  }
}

// ---------------- weight convert + transpose: Wt[n*K+k] = bf16(W[k*N+n]) -------
__global__ void fg_wconv(const float* __restrict__ W, bf16* __restrict__ Wt, int K, int N) {
  long i = (long)blockIdx.x * blockDim.x + threadIdx.x;
  long g = (long)gridDim.x * blockDim.x;
  long n_el = (long)K * N;
  for (; i < n_el; i += g) {
    int k = (int)(i / N), n = (int)(i % N);
    Wt[(long)n * K + k] = (bf16)W[i];
  }
}

// ---------------- TDM: DMA one 16xK bf16 A-panel into LDS ----------------
// D# group0: [1:0]=count=1, [63:32]=lds_addr, [120:64]=global_addr, [127:126]=type 2
// D# group1: [17:16]=data_size(1=2B), [79:48]=tensor_dim0=K, [111:80]=tensor_dim1=SMP,
//            [127:112]=tile_dim0=K, [143:128]=tile_dim1=16, [207:160]=dim0_stride=K
__device__ __forceinline__ void fg_tdm_load_panel(const bf16* gsrc, unsigned lds_off, int K) {
  unsigned long long ga = (unsigned long long)(__SIZE_TYPE__)gsrc;
  v4u g0;
  g0[0] = 1u;                                            // count=1
  g0[1] = lds_off;                                       // lds_addr
  g0[2] = (unsigned)(ga & 0xFFFFFFFFull);                // global_addr[31:0]
  g0[3] = (unsigned)((ga >> 32) & 0x01FFFFFFull) | (2u << 30);  // addr[56:32], type=2
  v8i g1;
  g1[0] = 1 << 16;                                       // data_size = 2 bytes
  g1[1] = (int)((K & 0xFFFF) << 16);                     // tensor_dim0[15:0]
  g1[2] = (int)(((SMP & 0xFFFF) << 16) | ((K >> 16) & 0xFFFF)); // dim1 lo | dim0 hi
  g1[3] = (int)(((K & 0xFFFF) << 16) | ((SMP >> 16) & 0xFFFF)); // tile_dim0 | dim1 hi
  g1[4] = 16;                                            // tile_dim1 = 16 rows
  g1[5] = K;                                             // tensor_dim0_stride[31:0]
  g1[6] = 0;
  g1[7] = 0;
  v4i gz = {0, 0, 0, 0};
#if defined(__clang_major__) && (__clang_major__ >= 23)
  v8i gz8 = {0, 0, 0, 0, 0, 0, 0, 0};
  __builtin_amdgcn_tensor_load_to_lds(g0, g1, gz, gz, gz8, 0);
#else
  __builtin_amdgcn_tensor_load_to_lds(g0, g1, gz, gz, 0);
#endif
}

// ---------------- WMMA GEMM: C[M,256] = A[M,K] x W[K,256] ----------------
// Block = 16 rows x all 256 cols. TDM DMAs the 16xK bf16 A-panel to LDS once;
// K-loop is fully unrolled ds_load_b128 + dual v_wmma per step. W is
// pre-transposed so each lane's 16 contiguous K-values are one 32B load.
template <int K>
__global__ __launch_bounds__(256) void fg_wmma_gemm(
    const bf16* __restrict__ Ab,     // [M, K] bf16 row-major
    const bf16* __restrict__ Wt,     // [256, K] transposed weights (bf16)
    float* __restrict__ C) {         // [M, 256]
  __shared__ alignas(32) bf16 As[16 * K];
  int tid  = threadIdx.x;
  int lane = tid & 31, wave = tid >> 5;
  int mrow = lane & 15, half = lane >> 4;
  int tileM = blockIdx.x;
  if (tid < 32) {
    const bf16* gsrc = Ab + (long)tileM * 16 * K;
    fg_tdm_load_panel(gsrc, (unsigned)(__SIZE_TYPE__)(void*)As, K);
    __builtin_amdgcn_s_wait_tensorcnt(0);
  }
  __syncthreads();
  v8f acc0 = {}, acc1 = {};
  int n0 = wave * 32;
  const v8bf* arow = (const v8bf*)(As + mrow * K);
  const bf16* b0row = Wt + (long)(n0 + mrow) * K + half * 16;
  const bf16* b1row = Wt + (long)(n0 + 16 + mrow) * K + half * 16;
#pragma unroll
  for (int kb = 0; kb < K; kb += 32) {
    int kc = kb >> 3;
    // A fragment: row mrow, K = kb+16*half..+15
    v16bf a = __builtin_shufflevector(arow[kc + 2 * half], arow[kc + 2 * half + 1],
                                      0, 1, 2, 3, 4, 5, 6, 7,
                                      8, 9, 10, 11, 12, 13, 14, 15);
    v16bf b0 = *(const v16bf*)(b0row + kb);
    v16bf b1 = *(const v16bf*)(b1row + kb);
    acc0 = __builtin_amdgcn_wmma_f32_16x16x32_bf16(false, a, false, b0,
                                                   (short)0, acc0, false, false);
    acc1 = __builtin_amdgcn_wmma_f32_16x16x32_bf16(false, a, false, b1,
                                                   (short)0, acc1, false, false);
  }
  // C layout: VGPR r -> M = r + 8*half, N = lane&15
  float* crow = C + (long)tileM * 16 * HD;
#pragma unroll
  for (int r = 0; r < 8; ++r) {
    int M = r + half * 8;
    crow[(long)M * HD + n0 + mrow]      = acc0[r];
    crow[(long)M * HD + n0 + 16 + mrow] = acc1[r];
  }
}

// ---------------- 256-wide scatter-add over valid edges ----------------
__global__ void fg_scatter256(const float* __restrict__ sup, float* __restrict__ dst,
                              const int* __restrict__ rowc, const int* __restrict__ colc,
                              const int* __restrict__ ec) {
  long total = (long)(*ec) << 6;  // 64 float4 chunks per edge
  long i = (long)blockIdx.x * blockDim.x + threadIdx.x;
  long g = (long)gridDim.x * blockDim.x;
  for (; i < total; i += g) {
    long e = i >> 6;
    int c = (int)(i & 63) * 4;
    int r = rowc[e], cl = colc[e];
    const float4 v = *(const float4*)(sup + (long)r * HD + c);
    float* d = dst + (long)cl * HD + c;
    atomicAdd(d + 0, v.x);
    atomicAdd(d + 1, v.y);
    atomicAdd(d + 2, v.z);
    atomicAdd(d + 3, v.w);
  }
}

// ---------------- BN stats: each thread owns channel c = tid ----------------
__global__ void fg_bn_stats(const float* __restrict__ h, float* __restrict__ stats) {
  int c = threadIdx.x;  // 256
  float s = 0.f, s2 = 0.f;
  for (int r = blockIdx.x; r < SMP; r += gridDim.x) {
    float v = h[(long)r * HD + c];
    s += v;
    s2 += v * v;
  }
  atomicAdd(&stats[c], s);
  atomicAdd(&stats[HD + c], s2);
}

// ---------------- BN + ReLU; mode 0 -> bf16 out, mode 1 -> f32 in place -------
__global__ void fg_bn_relu(const float* __restrict__ h, const float* __restrict__ stats,
                           const float* __restrict__ gamma, const float* __restrict__ beta,
                           bf16* __restrict__ outb, float* __restrict__ outf, int mode) {
  long n = (long)SMP * HD;
  long i = (long)blockIdx.x * blockDim.x + threadIdx.x;
  long g = (long)gridDim.x * blockDim.x;
  const float invS = 1.0f / (float)SMP;
  for (; i < n; i += g) {
    int c = (int)(i & (HD - 1));
    float mean = stats[c] * invS;
    float var = stats[HD + c] * invS - mean * mean;
    float y = gamma[c] * (h[i] - mean) * rsqrtf(var + BN_EPS) + beta[c];
    y = fmaxf(y, 0.0f);
    if (mode == 0) outb[i] = (bf16)y;
    else outf[i] = y;
  }
}

// ---------------- conv3 support: per-row dot with W3 (one wave / row) ---------
__global__ void fg_dot_w3(const float* __restrict__ h, const float* __restrict__ W3,
                          float* __restrict__ s3) {
  int wave = threadIdx.x >> 5, lane = threadIdx.x & 31;
  int row = blockIdx.x * 8 + wave;
  if (row >= SMP) return;
  float acc = 0.f;
  for (int k = lane; k < HD; k += 32) acc += h[(long)row * HD + k] * W3[k];
  for (int off = 16; off > 0; off >>= 1) acc += __shfl_down(acc, off, 32);
  if (lane == 0) s3[row] = acc;
}

// ---------------- out init with b3, then scalar scatter ----------------
__global__ void fg_init_out(float* __restrict__ out, const float* __restrict__ b3) {
  int i = blockIdx.x * blockDim.x + threadIdx.x;
  if (i < SMP) out[i] = b3[0];
}
__global__ void fg_scatter1(const float* __restrict__ s3, float* __restrict__ out,
                            const int* __restrict__ rowc, const int* __restrict__ colc,
                            const int* __restrict__ ec) {
  long total = (long)(*ec);
  long i = (long)blockIdx.x * blockDim.x + threadIdx.x;
  long g = (long)gridDim.x * blockDim.x;
  for (; i < total; i += g) atomicAdd(&out[colc[i]], s3[rowc[i]]);
}

// ================= host launcher =================
extern "C" void kernel_launch(void* const* d_in, const int* in_sizes, int n_in,
                              void* d_out, int out_size, void* d_ws, size_t ws_size,
                              hipStream_t stream) {
  const float* x   = (const float*)d_in[0];
  const int*   ei  = (const int*)d_in[1];   // [2, NE]: src then dst
  const float* W1  = (const float*)d_in[2];
  const float* W2  = (const float*)d_in[4];
  const float* W3  = (const float*)d_in[6];
  const float* b3  = (const float*)d_in[7];
  const float* g1  = (const float*)d_in[8];
  const float* be1 = (const float*)d_in[9];
  const float* g2  = (const float*)d_in[10];
  const float* be2 = (const float*)d_in[11];
  const int* src = ei;
  const int* dst = ei + NE;

  float* out     = (float*)d_out;            // [SMP] f32
  int*   sampled = ((int*)d_out) + SMP;      // [SMP] int32 (bit-cast region)

  // workspace carve-out (all 256B aligned); ~143 MB total
  char* ws = (char*)d_ws;
  size_t off = 0;
  auto take = [&](size_t bytes) -> void* {
    void* p = ws + off;
    off = (off + bytes + 255) & ~(size_t)255;
    return p;
  };
  float* deg    = (float*)take((size_t)NN * 4);
  float* cdf    = (float*)take((size_t)NN * 4);
  float* total  = (float*)take(4);
  int*   mapping= (int*)  take((size_t)NN * 4);
  int*   ec     = (int*)  take(4);
  int*   rowc   = (int*)  take((size_t)NE * 4);
  int*   colc   = (int*)  take((size_t)NE * 4);
  bf16*  xsb    = (bf16*) take((size_t)SMP * INF_ * 2);
  bf16*  w1t    = (bf16*) take((size_t)INF_ * HD * 2);
  bf16*  w2t    = (bf16*) take((size_t)HD * HD * 2);
  float* sup    = (float*)take((size_t)SMP * HD * 4);
  float* hbuf   = (float*)take((size_t)SMP * HD * 4);
  bf16*  hb     = (bf16*) take((size_t)SMP * HD * 2);
  float* stats  = (float*)take((size_t)2 * HD * 4);
  float* s3     = (float*)take((size_t)SMP * 4);
  (void)ws_size; (void)in_sizes; (void)n_in; (void)out_size;

  // ---- sampling pipeline ----
  fg_fill_f32<<<512, 256, 0, stream>>>(deg, NN, 0.0f);
  fg_fill_i32<<<512, 256, 0, stream>>>(mapping, NN, -1);
  fg_fill_i32<<<1, 32, 0, stream>>>(ec, 1, 0);
  fg_degree<<<4096, 256, 0, stream>>>(src, deg);
  fg_scan<<<1, 1024, 0, stream>>>(deg, cdf, total);
  fg_sample<<<SMP / 256, 256, 0, stream>>>(cdf, total, sampled);
  fg_mapping<<<SMP / 256, 256, 0, stream>>>(sampled, mapping);
  fg_compact<<<4096, 256, 0, stream>>>(src, dst, mapping, rowc, colc, ec);

  // ---- bf16 staging: gathered features + transposed weights ----
  fg_gather_x<<<4096, 256, 0, stream>>>(x, sampled, xsb);
  fg_wconv<<<512, 256, 0, stream>>>(W1, w1t, INF_, HD);
  fg_wconv<<<256, 256, 0, stream>>>(W2, w2t, HD, HD);

  // ---- layer 1: TDM+WMMA GEMM, scatter, BN+ReLU ----
  fg_wmma_gemm<INF_><<<SMP / 16, 256, 0, stream>>>(xsb, w1t, sup);
  fg_fill_f32<<<4096, 256, 0, stream>>>(hbuf, (long)SMP * HD, 0.0f);
  fg_scatter256<<<4096, 256, 0, stream>>>(sup, hbuf, rowc, colc, ec);
  fg_fill_f32<<<2, 256, 0, stream>>>(stats, 2 * HD, 0.0f);
  fg_bn_stats<<<128, 256, 0, stream>>>(hbuf, stats);
  fg_bn_relu<<<4096, 256, 0, stream>>>(hbuf, stats, g1, be1, hb, (float*)0, 0);

  // ---- layer 2 ----
  fg_wmma_gemm<HD><<<SMP / 16, 256, 0, stream>>>(hb, w2t, sup);
  fg_fill_f32<<<4096, 256, 0, stream>>>(hbuf, (long)SMP * HD, 0.0f);
  fg_scatter256<<<4096, 256, 0, stream>>>(sup, hbuf, rowc, colc, ec);
  fg_fill_f32<<<2, 256, 0, stream>>>(stats, 2 * HD, 0.0f);
  fg_bn_stats<<<128, 256, 0, stream>>>(hbuf, stats);
  fg_bn_relu<<<4096, 256, 0, stream>>>(hbuf, stats, g2, be2, (bf16*)0, hbuf, 1);

  // ---- layer 3: per-row dot, init with b3, scalar scatter ----
  fg_dot_w3<<<SMP / 8, 256, 0, stream>>>(hbuf, W3, s3);
  fg_init_out<<<SMP / 256, 256, 0, stream>>>(out, b3);
  fg_scatter1<<<1024, 256, 0, stream>>>(s3, out, rowc, colc, ec);
}
